// MultiHeadExpectation_80659485819177
// MI455X (gfx1250) — compile-verified, matrix operand order checked by reading, not run on hardware
//
#include <hip/hip_runtime.h>
#include <hip/hip_bf16.h>

// Problem constants (match reference)
#define NB 8
#define NU 512
#define NH 8
#define CH 64            // NU / NH
#define NX 1024
#define NY 1024
#define SCALEF 0.125f    // 1/sqrt(CH)

#define LDSROW 72        // 64 data halfs + 8 pad halfs (16B) per panel row

typedef __attribute__((ext_vector_type(16))) __bf16       v16bf;
typedef __attribute__((ext_vector_type(8)))  float        v8f;
typedef __attribute__((ext_vector_type(4)))  unsigned int v4u;
typedef __attribute__((ext_vector_type(4)))  int          v4i;
typedef __attribute__((ext_vector_type(8)))  int          v8i;

struct alignas(16) BFC { __bf16 v[8]; };          // one 16-byte bf16 chunk
union  AFrag { v16bf v; BFC c[2]; };

#if __has_builtin(__builtin_amdgcn_tensor_load_to_lds) && __has_builtin(__builtin_amdgcn_s_wait_tensorcnt)
#define MHE_TDM 1
#else
#define MHE_TDM 0
#endif

// ISA 7.12.2 16-bit A/B fragment: halfs 0..7 = k in [8g, 8g+7], halfs 8..15 =
// k in [16+8g, 16+8g+7] -> two contiguous 16B loads per lane per 32-k window.
__device__ __forceinline__ v16bf load_frag(const __bf16* krow, int g) {
    AFrag f;
    f.c[0] = *(const BFC*)(krow + 8 * g);
    f.c[1] = *(const BFC*)(krow + 16 + 8 * g);
    return f.v;
}

// ---------------------------------------------------------------------------
// Kernel 0: zero accumulators in workspace
// ---------------------------------------------------------------------------
__global__ void mhe_init(float* __restrict__ sim, int* __restrict__ nel) {
    int t = threadIdx.x;
    if (t < NB * NH) sim[t] = 0.0f;
    if (t < NB) nel[t] = 0;
}

// ---------------------------------------------------------------------------
// Kernel 1: W f32 -> bf16 (row-major, coalesced)
// ---------------------------------------------------------------------------
__global__ __launch_bounds__(256) void mhe_cvtW(const float* __restrict__ W,
                                                __bf16* __restrict__ Wb) {
    int i = (blockIdx.x * blockDim.x + threadIdx.x) * 4;
    #pragma unroll
    for (int j = 0; j < 4; ++j) Wb[i + j] = (__bf16)W[i + j];
}

// ---------------------------------------------------------------------------
// Kernel 2: transpose+convert x,y [B][U][N] f32 -> xT,yT [B][N][U] bf16
// 32x32 LDS tiles, coalesced on both sides
// ---------------------------------------------------------------------------
__global__ __launch_bounds__(256) void mhe_transpose(
    const float* __restrict__ x, const float* __restrict__ y,
    __bf16* __restrict__ xT, __bf16* __restrict__ yT)
{
    __shared__ float tile[32][33];
    int blk = blockIdx.x;
    int nt = blk & 31; blk >>= 5;   // 32 n-tiles
    int ut = blk & 15; blk >>= 4;   // 16 u-tiles
    int b  = blk & 7;  blk >>= 3;
    int which = blk;                // 0 = x, 1 = y
    const float* src = which ? y : x;
    __bf16*      dst = which ? yT : xT;

    int r = threadIdx.x >> 5, c = threadIdx.x & 31;
    int u0 = ut * 32, n0 = nt * 32;
    #pragma unroll
    for (int rr = 0; rr < 4; ++rr) {
        int row = r + 8 * rr;
        tile[row][c] = src[(size_t)b * NU * NX + (size_t)(u0 + row) * NX + n0 + c];
    }
    __syncthreads();
    #pragma unroll
    for (int rr = 0; rr < 4; ++rr) {
        int row = r + 8 * rr;
        dst[(size_t)b * NX * NU + (size_t)(n0 + row) * NU + u0 + c] = (__bf16)tile[c][row];
    }
}

// ---------------------------------------------------------------------------
// Kernel 3: projection GEMM (bf16 WMMA, f32 acc), 2x2 register blocking.
// One wave = 32x32 output tile: 2 A-frags (W rows) x 2 B-frags (xT/yT rows)
// -> 4 WMMAs per 32-k step, 16 FLOP/byte from L2.
// Output stored transposed: XpT/YpT [b][n][u] bf16.
// ---------------------------------------------------------------------------
__global__ __launch_bounds__(256) void mhe_proj(
    const __bf16* __restrict__ Wb, const __bf16* __restrict__ xT,
    const __bf16* __restrict__ yT, __bf16* __restrict__ XpT,
    __bf16* __restrict__ YpT)
{
    int lane = threadIdx.x & 31;
    int wave = (int)((blockIdx.x * blockDim.x + threadIdx.x) >> 5);
    int ng = wave & 31; wave >>= 5;   // 32 groups of 2 n-tiles
    int mg = wave & 15; wave >>= 4;   // 16 groups of 2 m-tiles (out channels)
    int b  = wave & 7;  wave >>= 3;
    int which = wave;                 // 0 = x, 1 = y

    const __bf16* src = which ? yT : xT;
    __bf16*       dst = which ? YpT : XpT;

    int m0 = mg * 32, n0 = ng * 32;
    int lm = lane & 15, g = lane >> 4;

    const __bf16* wrow0 = Wb  + (size_t)(m0 + lm) * NU;
    const __bf16* wrow1 = wrow0 + (size_t)16 * NU;
    const __bf16* xrow0 = src + (size_t)b * NX * NU + (size_t)(n0 + lm) * NU;
    const __bf16* xrow1 = xrow0 + (size_t)16 * NU;

    v8f c00 = {}, c01 = {}, c10 = {}, c11 = {};
    #pragma unroll 2
    for (int kk = 0; kk < NU; kk += 32) {
        v16bf a0 = load_frag(wrow0 + kk, g);
        v16bf a1 = load_frag(wrow1 + kk, g);
        v16bf b0 = load_frag(xrow0 + kk, g);
        v16bf b1 = load_frag(xrow1 + kk, g);
        c00 = __builtin_amdgcn_wmma_f32_16x16x32_bf16(false, a0, false, b0,
                                                      (short)0, c00, false, false);
        c01 = __builtin_amdgcn_wmma_f32_16x16x32_bf16(false, a0, false, b1,
                                                      (short)0, c01, false, false);
        c10 = __builtin_amdgcn_wmma_f32_16x16x32_bf16(false, a1, false, b0,
                                                      (short)0, c10, false, false);
        c11 = __builtin_amdgcn_wmma_f32_16x16x32_bf16(false, a1, false, b1,
                                                      (short)0, c11, false, false);
    }

    // D tiles: lane holds (o = m_base + r + 8g, n = n_base + lm);
    // 8 contiguous channels -> one b128 store per sub-tile.
    v8f cs[4] = { c00, c01, c10, c11 };
    #pragma unroll
    for (int ti = 0; ti < 2; ++ti) {          // m sub-tile
        #pragma unroll
        for (int tj = 0; tj < 2; ++tj) {      // n sub-tile
            BFC o;
            v8f cv = cs[ti * 2 + tj];
            #pragma unroll
            for (int r = 0; r < 8; ++r) o.v[r] = (__bf16)cv[r];
            *(BFC*)(dst + (size_t)b * NX * NU
                        + (size_t)(n0 + tj * 16 + lm) * NU
                        + (m0 + ti * 16) + 8 * g) = o;
        }
    }
}

// ---------------------------------------------------------------------------
// Kernel 4: n_el[b] = popcount(mask[b])  (mask = 1-byte bool)
// ---------------------------------------------------------------------------
__global__ __launch_bounds__(256) void mhe_nel(
    const unsigned char* __restrict__ mask, int* __restrict__ nel)
{
    size_t tid  = (size_t)blockIdx.x * blockDim.x + threadIdx.x;
    size_t base = tid * 16;
    int b = (int)(base >> 20);              // 1 MB per batch plane
    const uchar4* p = (const uchar4*)(mask + base);
    int s = 0;
    #pragma unroll
    for (int i = 0; i < 4; ++i) {
        uchar4 v = p[i];
        s += (v.x != 0) + (v.y != 0) + (v.z != 0) + (v.w != 0);
    }
    #pragma unroll
    for (int off = 16; off > 0; off >>= 1) s += __shfl_xor(s, off, 32);
    if ((threadIdx.x & 31) == 0) atomicAdd(&nel[b], s);
}

// ---------------------------------------------------------------------------
// Kernel 5: fused masked-relu score reduction.
// Block = one (b, h, 256-col m-panel); the 256x64 bf16 Y-panel (32 KB) is
// DMA'd into LDS once by the Tensor Data Mover and reused by all 8 waves x
// 16 m-tiles. Score tiles live only in the WMMA accumulator.
// ---------------------------------------------------------------------------
__global__ __launch_bounds__(256) void mhe_score(
    const __bf16* __restrict__ XpT, const __bf16* __restrict__ YpT,
    const unsigned char* __restrict__ mask, float* __restrict__ sim)
{
    __shared__ alignas(16) __bf16 ypanel[256 * LDSROW];   // 36 KB, padded rows

    int blk = blockIdx.x;
    int ntg = blk & 7; blk >>= 3;   // 8 n-tile groups (8 waves each)
    int mg  = blk & 3; blk >>= 2;   // 4 m-panels of 256 columns
    int h   = blk & 7; blk >>= 3;
    int b   = blk;

    int wid = threadIdx.x >> 5, lane = threadIdx.x & 31;
    int lm = lane & 15, g = lane >> 4;
    int m0p = mg * 256;

    const __bf16* ysrc = YpT + (size_t)b * NY * NU + (size_t)m0p * NU + h * CH;

#if MHE_TDM
    if (threadIdx.x < 32) {
        // Tensor DMA descriptor (ISA 8.3/8.4): 2D tile 64 x 256 bf16,
        // row stride NU elements, LDS pad 16B per 128B row.
        unsigned long long ga = (unsigned long long)(size_t)ysrc;
        unsigned int ldsoff = (unsigned int)(size_t)&ypanel[0];
        v4u g0 = { 1u,                                   // count=1, user mode
                   ldsoff,                               // lds_addr
                   (unsigned int)ga,                     // global_addr[31:0]
                   (unsigned int)((ga >> 32) & 0x1FFFFFFu) | (2u << 30) }; // +type=2
        v8i g1 = { (int)((1u << 16) | (1u << 20) | (4u << 22) | (3u << 25)),
                   // data_size=2B, pad_enable, pad_interval=128B, pad_amount=16B
                   0,                                    // td0[15:0]<<16 (td0=2^20)
                   0x10,                                 // td0[31:16] | td1[15:0]<<16
                   0x10 | (64 << 16),                    // td1[31:16] | tile_dim0=64
                   256,                                  // tile_dim1=256, tile_dim2=0
                   512,                                  // tensor_dim0_stride = NU
                   0, 0 };
        v4i gz = { 0, 0, 0, 0 };
#if defined(__clang_major__) && __clang_major__ >= 23
        v8i gz8 = { 0, 0, 0, 0, 0, 0, 0, 0 };
        __builtin_amdgcn_tensor_load_to_lds(g0, g1, gz, gz, gz8, 0);
#else
        __builtin_amdgcn_tensor_load_to_lds(g0, g1, gz, gz, 0);
#endif
        __builtin_amdgcn_s_wait_tensorcnt(0);
    }
#else
    // Fallback: cooperative 16B-chunk copy global -> padded LDS
    for (int cidx = threadIdx.x; cidx < 2048; cidx += 256) {
        int row = cidx >> 3, ch = cidx & 7;
        *(BFC*)(ypanel + row * LDSROW + ch * 8) =
            *(const BFC*)(ysrc + (size_t)row * NU + ch * 8);
    }
#endif
    __syncthreads();

    // A fragments (rows = n, K = head channels), contiguous b128 from XpT
    int nt = ntg * 8 + wid;
    int n  = nt * 16 + lm;
    const __bf16* arow = XpT + (size_t)b * NX * NU + (size_t)n * NU + h * CH;
    v16bf a0 = load_frag(arow, g);
    v16bf a1 = load_frag(arow + 32, g);

    float acc = 0.0f;
    const unsigned char* mb = mask + (size_t)b * NX * NY;

    for (int t = 0; t < 16; ++t) {
        int lrow = t * 16 + lm;
        const __bf16* brow = ypanel + lrow * LDSROW;
        v16bf b0 = load_frag(brow, g);
        v16bf b1 = load_frag(brow + 32, g);
        v8f d = {};
        d = __builtin_amdgcn_wmma_f32_16x16x32_bf16(false, a0, false, b0,
                                                    (short)0, d, false, false);
        d = __builtin_amdgcn_wmma_f32_16x16x32_bf16(false, a1, false, b1,
                                                    (short)0, d, false, false);
        int me = m0p + t * 16 + lm;
        #pragma unroll
        for (int r = 0; r < 8; ++r) {
            int ne = nt * 16 + r + 8 * g;
            float v = d[r] > 0.0f ? d[r] : 0.0f;
            if (mb[(size_t)ne * NY + me]) acc += v;
        }
    }

    #pragma unroll
    for (int off = 16; off > 0; off >>= 1) acc += __shfl_xor(acc, off, 32);
    if (lane == 0) atomicAdd(&sim[b * NH + h], acc);
}

// ---------------------------------------------------------------------------
// Kernel 6: out[b] = sum_h (sim[b,h]*SCALE/n_el[b]) * fc_w[h] + fc_b
// ---------------------------------------------------------------------------
__global__ void mhe_final(const float* __restrict__ sim, const int* __restrict__ nel,
                          const float* __restrict__ fcw, const float* __restrict__ fcb,
                          float* __restrict__ out)
{
    int b = threadIdx.x;
    if (b < NB) {
        int nz = nel[b];
        float ne = (float)(nz == 0 ? 1 : nz);
        float s = 0.0f;
        #pragma unroll
        for (int hh = 0; hh < NH; ++hh) s += sim[b * NH + hh] * fcw[hh];
        out[b] = s * SCALEF / ne + fcb[0];
    }
}

// ---------------------------------------------------------------------------
extern "C" void kernel_launch(void* const* d_in, const int* in_sizes, int n_in,
                              void* d_out, int out_size, void* d_ws, size_t ws_size,
                              hipStream_t stream) {
    const float*         x    = (const float*)d_in[0];
    const float*         y    = (const float*)d_in[1];
    const unsigned char* mask = (const unsigned char*)d_in[2];   // jax bool = 1 byte
    const float*         W    = (const float*)d_in[3];
    const float*         fcw  = (const float*)d_in[4];
    const float*         fcb  = (const float*)d_in[5];
    float*               out  = (float*)d_out;

    char*   ws  = (char*)d_ws;
    const size_t MB = 1024ull * 1024ull;
    __bf16* XpT = (__bf16*)(ws);                 // 8 MB  [B][NX][NU]
    __bf16* YpT = (__bf16*)(ws +  8 * MB);       // 8 MB  [B][NY][NU]
    __bf16* xT  = (__bf16*)(ws + 16 * MB);       // 8 MB  [B][NX][NU]
    __bf16* yT  = (__bf16*)(ws + 24 * MB);       // 8 MB  [B][NY][NU]
    __bf16* Wb  = (__bf16*)(ws + 32 * MB);       // 512 KB
    float*  sim = (float*) (ws + 32 * MB + 512 * 1024);
    int*    nel = (int*)   (ws + 32 * MB + 512 * 1024 + 256);

    mhe_init<<<1, 128, 0, stream>>>(sim, nel);
    // 512*512 / (256*4) = 256 blocks
    mhe_cvtW<<<256, 256, 0, stream>>>(W, Wb);
    // 2 tensors * 8 b * 16 u-tiles * 32 n-tiles = 8192 blocks
    mhe_transpose<<<8192, 256, 0, stream>>>(x, y, xT, yT);
    // waves = 2 * 8 * 16 * 32 = 8192 -> 1024 blocks (32x32 tile per wave)
    mhe_proj<<<1024, 256, 0, stream>>>(Wb, xT, yT, XpT, YpT);
    // 8 MB of mask / 16 B per thread -> 2048 blocks
    mhe_nel<<<2048, 256, 0, stream>>>(mask, nel);
    // blocks = 8 b * 8 h * 4 m-panels * 8 n-groups = 2048
    mhe_score<<<2048, 256, 0, stream>>>(XpT, YpT, mask, sim);
    mhe_final<<<1, 32, 0, stream>>>(sim, nel, fcw, fcb, out);
}
